// ConvLAG_15814069584290
// MI455X (gfx1250) — compile-verified, matrix operand order checked by reading, not run on hardware
//
#include <hip/hip_runtime.h>
#include <hip/hip_bf16.h>
#include <math.h>

// ---------------------------------------------------------------------------
// ConvLAG attention for MI455X (gfx1250, wave32, WMMA bf16 16x16x32).
//
// Phase 1: project qq->qB, kv->kB (natural [tok][ch][spat] bf16) and kv->vT
//          (per-head transposed [b,n][s=4096][t=128] bf16) into d_ws.
//          160 MB total: fits the 192 MB global L2, so phase 2 reads from L2.
// Phase 2: per (b, head): S = Q@K^T (bf16 WMMA, fp32 accum), softmax,
//          O = P@V via LDS-staged P and transposed V. Writes out + sim.
// ---------------------------------------------------------------------------

typedef __attribute__((ext_vector_type(16))) __bf16        v16bf;
typedef __attribute__((ext_vector_type(8)))  float         v8f;
typedef __attribute__((ext_vector_type(4)))  unsigned int  u32x4;

union Frag {
    v16bf bf;
    u32x4 u[2];   // two 16-byte halves: K = base8+{0..7} and K = 16+base8+{0..7}
};

#define TOKEN_ELEMS 16384   // 64 ch * 256 spatial
#define NQ_TOK      1024    // B*Tq
#define NKV_TOK     2048    // B*T

static __device__ __forceinline__ v8f wmma_bf16(const Frag& a, const Frag& b, v8f c) {
    return __builtin_amdgcn_wmma_f32_16x16x32_bf16(
        /*neg_a=*/false, a.bf, /*neg_b=*/false, b.bf,
        /*c_mod=*/(short)0, c, /*reuse_a=*/false, /*reuse_b=*/false);
}

// GEMM core for one N-tile: acc[mt] += W(64x64) @ X(64x16-slice), K=64 (2x32).
static __device__ __forceinline__ void proj_tile(
    const __bf16 (*__restrict__ Wl)[72], const __bf16 (*__restrict__ Xl)[72],
    int nt, int l16, int base8, v8f acc[4])
{
    #pragma unroll
    for (int ks = 0; ks < 2; ++ks) {
        Frag B;
        const __bf16* pb = &Xl[nt * 16 + l16][ks * 32 + base8];
        B.u[0] = *(const u32x4*)pb;
        B.u[1] = *(const u32x4*)(pb + 16);
        #pragma unroll
        for (int mt = 0; mt < 4; ++mt) {
            Frag A;
            const __bf16* pa = &Wl[mt * 16 + l16][ks * 32 + base8];
            A.u[0] = *(const u32x4*)pa;
            A.u[1] = *(const u32x4*)(pa + 16);
            acc[mt] = wmma_bf16(A, B, acc[mt]);
        }
    }
}

// Cooperative loads: W (64x64 f32 -> bf16, row-major) and X^T (f32 -> bf16).
static __device__ __forceinline__ void load_W(
    const float* __restrict__ Wsrc, __bf16 (*__restrict__ Wl)[72], int tid)
{
    for (int i = tid; i < 4096; i += 256)
        Wl[i >> 6][i & 63] = (__bf16)Wsrc[i];
}
static __device__ __forceinline__ void load_XT(
    const float* __restrict__ X, __bf16 (*__restrict__ Xl)[72], int tid)
{
    const float* xc = X + tid;               // column s = tid, walk channels
    #pragma unroll 8
    for (int c = 0; c < 64; ++c)
        Xl[tid][c] = (__bf16)xc[c * 256];    // coalesced across threads per c
}

// ===================== Kernel 1a: Q projection (1024 WGs) ==================
__global__ __launch_bounds__(256) void convlag_proj_q_kernel(
    const float* __restrict__ qq,
    const float* __restrict__ Wq, const float* __restrict__ bq,
    __bf16* __restrict__ qB)
{
    __shared__ __bf16 Wlds[64][72];
    __shared__ __bf16 Xlds[256][72];
    __shared__ float  blds[64];

    const int tid = threadIdx.x;
    const int tok = blockIdx.x;

    load_W(Wq, Wlds, tid);
    if (tid < 64) blds[tid] = bq[tid];
    load_XT(qq + (size_t)tok * TOKEN_ELEMS, Xlds, tid);
    __syncthreads();

    const int wave  = tid >> 5;
    const int lane  = tid & 31;
    const int l16   = lane & 15;
    const int lhalf = lane >> 4;
    const int base8 = lhalf * 8;

    #pragma unroll
    for (int nti = 0; nti < 2; ++nti) {
        const int nt = wave * 2 + nti;
        v8f acc[4] = {};
        proj_tile(Wlds, Xlds, nt, l16, base8, acc);
        #pragma unroll
        for (int mt = 0; mt < 4; ++mt) {
            const int o0 = mt * 16 + lhalf * 8;
            __bf16* dst = qB + (size_t)tok * TOKEN_ELEMS + o0 * 256 + nt * 16 + l16;
            const float* bp = &blds[o0];
            #pragma unroll
            for (int r = 0; r < 8; ++r)
                dst[r * 256] = (__bf16)(acc[mt][r] + bp[r]);
        }
    }
}

// ===================== Kernel 1b: K+V projection (2048 WGs) ================
__global__ __launch_bounds__(256) void convlag_proj_kv_kernel(
    const float* __restrict__ kv,
    const float* __restrict__ Wk, const float* __restrict__ bk,
    const float* __restrict__ Wv, const float* __restrict__ bv,
    __bf16* __restrict__ kB, __bf16* __restrict__ vT)
{
    __shared__ __bf16 Wlds[2][64][72];
    __shared__ __bf16 Xlds[256][72];
    __shared__ float  blds[2][64];

    const int tid = threadIdx.x;
    const int tok = blockIdx.x;              // 0..2047 = b*128 + t

    load_W(Wk, Wlds[0], tid);
    load_W(Wv, Wlds[1], tid);
    if (tid < 64) { blds[0][tid] = bk[tid]; blds[1][tid] = bv[tid]; }
    load_XT(kv + (size_t)tok * TOKEN_ELEMS, Xlds, tid);
    __syncthreads();

    const int wave  = tid >> 5;
    const int lane  = tid & 31;
    const int l16   = lane & 15;
    const int lhalf = lane >> 4;
    const int base8 = lhalf * 8;

    // ---- K: natural [tok][ch][spat] layout ----
    #pragma unroll
    for (int nti = 0; nti < 2; ++nti) {
        const int nt = wave * 2 + nti;
        v8f acc[4] = {};
        proj_tile(Wlds[0], Xlds, nt, l16, base8, acc);
        #pragma unroll
        for (int mt = 0; mt < 4; ++mt) {
            const int o0 = mt * 16 + lhalf * 8;
            __bf16* dst = kB + (size_t)tok * TOKEN_ELEMS + o0 * 256 + nt * 16 + l16;
            const float* bp = &blds[0][o0];
            #pragma unroll
            for (int r = 0; r < 8; ++r)
                dst[r * 256] = (__bf16)(acc[mt][r] + bp[r]);
        }
    }

    // ---- V: per-head transposed [b*4+n][srow=0..4095][t=0..127] ----
    // Head n == mt (16-channel heads align with 16-row M-tiles).
    const int b_ = tok >> 7, t = tok & 127;
    #pragma unroll
    for (int nti = 0; nti < 2; ++nti) {
        const int nt = wave * 2 + nti;
        v8f acc[4] = {};
        proj_tile(Wlds[1], Xlds, nt, l16, base8, acc);
        const int s = nt * 16 + l16;         // spatial 0..255
        #pragma unroll
        for (int mt = 0; mt < 4; ++mt) {
            __bf16* dst = vT + (((size_t)(b_ * 4 + mt)) * 4096
                                + (lhalf * 8) * 256 + s) * 128 + t;
            const float* bp = &blds[1][mt * 16 + lhalf * 8];
            #pragma unroll
            for (int r = 0; r < 8; ++r)
                dst[r * 32768] = (__bf16)(acc[mt][r] + bp[r]);   // srow += 256
        }
    }
}

// ============================ Kernel 2: attention ==========================
// One workgroup per (b, head). 8 waves.
__global__ __launch_bounds__(256) void convlag_attn_kernel(
    const __bf16* __restrict__ qB, const __bf16* __restrict__ kB,
    const __bf16* __restrict__ vT,
    float* __restrict__ out, float* __restrict__ simOut)
{
    __shared__ float  Slds[64][132];     // logits -> exp values, fp32
    __shared__ __bf16 Plds[64][136];     // probs, bf16 (padded, 16B-aligned rows)
    __shared__ float  invlds[64];        // per-row 1/sum

    const int wg = blockIdx.x;           // b*4 + n
    const int b  = wg >> 2, n = wg & 3;
    const int tid   = threadIdx.x;
    const int wave  = tid >> 5;
    const int lane  = tid & 31;
    const int l16   = lane & 15;
    const int lhalf = lane >> 4;
    const int base8 = lhalf * 8;

    // ---- Step A: S = Q(64x4096) @ K^T(4096x128); wave w owns N-tile w ----
    {
        v8f acc[4] = {};
        const __bf16* kp = kB + (size_t)(b * 128 + wave * 16 + l16) * TOKEN_ELEMS
                              + n * 4096 + base8;
        const __bf16* qbase = qB + (size_t)(b * 64 + l16) * TOKEN_ELEMS
                                 + n * 4096 + base8;
        const __bf16* ap0 = qbase;
        const __bf16* ap1 = qbase + (size_t)16 * TOKEN_ELEMS;
        const __bf16* ap2 = qbase + (size_t)32 * TOKEN_ELEMS;
        const __bf16* ap3 = qbase + (size_t)48 * TOKEN_ELEMS;
        for (int it = 0; it < 128; ++it) {
            Frag B, A0, A1, A2, A3;
            B.u[0]  = *(const u32x4*)kp;        B.u[1]  = *(const u32x4*)(kp + 16);
            A0.u[0] = *(const u32x4*)ap0;       A0.u[1] = *(const u32x4*)(ap0 + 16);
            A1.u[0] = *(const u32x4*)ap1;       A1.u[1] = *(const u32x4*)(ap1 + 16);
            A2.u[0] = *(const u32x4*)ap2;       A2.u[1] = *(const u32x4*)(ap2 + 16);
            A3.u[0] = *(const u32x4*)ap3;       A3.u[1] = *(const u32x4*)(ap3 + 16);
            acc[0] = wmma_bf16(A0, B, acc[0]);
            acc[1] = wmma_bf16(A1, B, acc[1]);
            acc[2] = wmma_bf16(A2, B, acc[2]);
            acc[3] = wmma_bf16(A3, B, acc[3]);
            kp += 32; ap0 += 32; ap1 += 32; ap2 += 32; ap3 += 32;
        }
        const float scale = 0.015625f;   // 1/sqrt(16*16*16) = 1/64
        #pragma unroll
        for (int mt = 0; mt < 4; ++mt)
            #pragma unroll
            for (int r = 0; r < 8; ++r)
                Slds[mt * 16 + lhalf * 8 + r][wave * 16 + l16] = acc[mt][r] * scale;
    }
    __syncthreads();

    // ---- Step B1: rowwise max/exp/sum (64 rows of 128) ----
    if (tid < 64) {
        float mx = -1e30f;
        for (int t = 0; t < 128; ++t) mx = fmaxf(mx, Slds[tid][t]);
        float sum = 0.f;
        for (int t = 0; t < 128; ++t) {
            const float e = __expf(Slds[tid][t] - mx);
            Slds[tid][t] = e;
            sum += e;
        }
        invlds[tid] = 1.f / sum;
    }
    __syncthreads();

    // ---- Step B2: normalize; emit sim (fp32, coalesced) and P (bf16) ----
    {
        float* simBase = simOut + (size_t)wg * 8192;
        for (int idx = tid; idx < 8192; idx += 256) {
            const int row = idx >> 7, t = idx & 127;
            const float p = Slds[row][t] * invlds[row];
            simBase[idx] = p;
            Plds[row][t] = (__bf16)p;
        }
    }
    __syncthreads();

    // ---- Step C: O = P(64x128) @ V(128x4096); wave w owns nt = w + 8*i ----
    for (int nti = 0; nti < 32; ++nti) {
        const int nt = wave + nti * 8;   // N-tile over 4096 (ch',spat) columns
        v8f acc[4] = {};
        const __bf16* vcol = vT + ((size_t)wg * 4096 + nt * 16 + l16) * 128 + base8;
        #pragma unroll
        for (int ks = 0; ks < 4; ++ks) {
            Frag B;
            B.u[0] = *(const u32x4*)(vcol + ks * 32);
            B.u[1] = *(const u32x4*)(vcol + ks * 32 + 16);
            #pragma unroll
            for (int mt = 0; mt < 4; ++mt) {
                Frag A;
                const __bf16* pa = &Plds[mt * 16 + l16][ks * 32 + base8];
                A.u[0] = *(const u32x4*)pa;
                A.u[1] = *(const u32x4*)(pa + 16);
                acc[mt] = wmma_bf16(A, B, acc[mt]);
            }
        }
        const int chp  = nt >> 4;                 // head-local channel 0..15
        const int spat = (nt & 15) * 16 + l16;    // spatial 0..255
        #pragma unroll
        for (int mt = 0; mt < 4; ++mt) {
            float* dst = out + (size_t)(b * 64 + mt * 16 + lhalf * 8) * TOKEN_ELEMS
                             + (n * 16 + chp) * 256 + spat;
            #pragma unroll
            for (int r = 0; r < 8; ++r)
                dst[r * TOKEN_ELEMS] = acc[mt][r];
        }
    }
}

// ================================ Launch ===================================
extern "C" void kernel_launch(void* const* d_in, const int* in_sizes, int n_in,
                              void* d_out, int out_size, void* d_ws, size_t ws_size,
                              hipStream_t stream) {
    const float* qq = (const float*)d_in[0];
    const float* kv = (const float*)d_in[1];
    const float* Wq = (const float*)d_in[2];
    const float* bq = (const float*)d_in[3];
    const float* Wk = (const float*)d_in[4];
    const float* bk = (const float*)d_in[5];
    const float* Wv = (const float*)d_in[6];
    const float* bv = (const float*)d_in[7];

    // Workspace layout (bf16): qB 32MB | kB 64MB | vT 64MB  -> 160MB, L2-resident.
    __bf16* qB = (__bf16*)d_ws;                                    // 1024*16384
    __bf16* kB = qB + (size_t)NQ_TOK  * TOKEN_ELEMS;               // 2048*16384
    __bf16* vT = kB + (size_t)NKV_TOK * TOKEN_ELEMS;               // 2048*16384

    float* out    = (float*)d_out;
    float* simOut = out + (size_t)NQ_TOK * TOKEN_ELEMS;            // 16*4*64*128

    convlag_proj_q_kernel<<<NQ_TOK, 256, 0, stream>>>(qq, Wq, bq, qB);
    convlag_proj_kv_kernel<<<NKV_TOK, 256, 0, stream>>>(kv, Wk, bk, Wv, bv, kB, vT);
    convlag_attn_kernel<<<64, 256, 0, stream>>>(qB, kB, vT, out, simOut);
}
